// SelfAttention_36807869727248
// MI455X (gfx1250) — matrix operand order repaired and checked by reading, not compiled, not run on hardware
//
#include <hip/hip_runtime.h>

// ---------------------------------------------------------------------------
// Self-attention for MI455X (gfx1250, wave32, WMMA 16x16x32 f16 -> f32 acc)
//   B=4, S=4096, D=256, fp32 in/out. Flash-attention streaming, f16 matrix
//   inputs with f32 accumulation. K-tile staging uses CDNA5 async
//   global->LDS loads (ASYNCcnt) when the toolchain exposes the builtins.
// ---------------------------------------------------------------------------

typedef __attribute__((ext_vector_type(16))) _Float16 v16h;
typedef __attribute__((ext_vector_type(8)))  _Float16 v8h;
typedef __attribute__((ext_vector_type(4)))  _Float16 v4h;
typedef __attribute__((ext_vector_type(8)))  float    v8f;
typedef __attribute__((ext_vector_type(4)))  int      v4i;

#define BB 4
#define SS 4096
#define DD 256

// LDS row strides (in halves); multiples of 8 halves (16B) for aligned v8h
// fragment access, padded to dodge bank conflicts.
#define KSTR 264   // K tile: 64 keys x 256 d (row-major: B-frags for Q*K^T)
#define VSTR 72    // V tile transposed: 256 d x 64 keys (B-frags for P*V)
#define PSTR 72    // per-wave P scratch: 16 rows x 64 keys
#define XSTR 264   // x tile: 64 rows x 256 d

// --- CDNA5 async global->LDS loads (GLOBAL_LOAD_ASYNC_TO_LDS_B128) --------
// Probe result (round 2 diagnostic): builtin exists with signature
//   void(__attribute__((address_space(1))) v4i*,   // global src
//        __attribute__((address_space(3))) v4i*,   // LDS dst
//        imm int offset, imm int cpol)
#if defined(__has_builtin)
#if __has_builtin(__builtin_amdgcn_global_load_async_to_lds_b128)
#define HAVE_ASYNC_LDS 1
#endif
#endif

#ifdef HAVE_ASYNC_LDS
typedef __attribute__((address_space(1))) v4i gv4i;
typedef __attribute__((address_space(3))) v4i lv4i;
__device__ inline void async_ld_b128(const _Float16* gsrc, _Float16* ldst) {
  __builtin_amdgcn_global_load_async_to_lds_b128(
      (gv4i*)gsrc, (lv4i*)ldst, 0, 0);
}
__device__ inline void wait_async0() {
#if __has_builtin(__builtin_amdgcn_s_wait_asynccnt)
  __builtin_amdgcn_s_wait_asynccnt(0);
#else
  asm volatile("s_wait_asynccnt 0x0" ::: "memory");
#endif
}
#endif

// ---- WMMA fragment helpers (K contiguous per lane half in A and B, not the ISA table order) -----

// A-matrix 16x32 f16: lane(0..15)=row M, half-group h=lane>>4 selects
// K in [16h,16h+16) contiguous (matches load_b_frag).
__device__ inline v16h load_a_frag(const _Float16* base, int stride, int k0) {
  const int lane = threadIdx.x & 31;
  const int h = lane >> 4;
  const int r = lane & 15;
  const _Float16* p = base + (size_t)r * stride + k0 + 16 * h;
  v8h lo = *(const v8h*)p;         // K = k0+16h .. +8
  v8h hi = *(const v8h*)(p + 8);   // K = k0+16h+8 .. +8
  v16h a;
#pragma unroll
  for (int i = 0; i < 8; ++i) { a[i] = lo[i]; a[i + 8] = hi[i]; }
  return a;
}

// B-matrix 32x16 f16: lane(0..15)=col N, lanes 16-31 same cols with K+16;
// each lane holds 16 contiguous K. Source layout: element(k,n) at
// base[n*stride + k] ("n-major"), so lanes read 32 contiguous bytes.
__device__ inline v16h load_b_frag(const _Float16* base, int stride, int k0) {
  const int lane = threadIdx.x & 31;
  const int h = lane >> 4;
  const int r = lane & 15;
  const _Float16* p = base + (size_t)r * stride + k0 + 16 * h;
  v8h lo = *(const v8h*)p;
  v8h hi = *(const v8h*)(p + 8);
  v16h b;
#pragma unroll
  for (int i = 0; i < 8; ++i) { b[i] = lo[i]; b[i + 8] = hi[i]; }
  return b;
}

__device__ inline v8f wmma16(v16h a, v16h b, v8f c) {
  return __builtin_amdgcn_wmma_f32_16x16x32_f16(
      /*neg_a=*/false, a, /*neg_b=*/false, b,
      /*c_mod=*/(short)0, c, /*reuse_a=*/false, /*reuse_b=*/false);
}

__device__ inline v8f vzero8() {
  v8f z = {0.f, 0.f, 0.f, 0.f, 0.f, 0.f, 0.f, 0.f};
  return z;
}

// ---- Kernel 0: convert weights fp32 -> f16 (layout [n][k] kept: already
// the B-fragment layout for y = x @ W^T) ------------------------------------
__global__ void __launch_bounds__(256) cvt_weights(
    const float* __restrict__ Wq, const float* __restrict__ Wk,
    const float* __restrict__ Wv, _Float16* __restrict__ Wqh,
    _Float16* __restrict__ Wkh, _Float16* __restrict__ Wvh) {
  const int i = blockIdx.x * 256 + threadIdx.x;
  if (i < DD * DD) {
    Wqh[i] = (_Float16)Wq[i];
    Wkh[i] = (_Float16)Wk[i];
    Wvh[i] = (_Float16)Wv[i];
  }
}

// ---- Kernel 1: QKV projection. 64 rows/block, 8 waves x 32 cols -----------
__global__ void __launch_bounds__(256) qkv_proj(
    const float* __restrict__ x,
    const _Float16* __restrict__ Wqh, const _Float16* __restrict__ Wkh,
    const _Float16* __restrict__ Wvh,
    const float* __restrict__ bq, const float* __restrict__ bk,
    const float* __restrict__ bv,
    _Float16* __restrict__ Qh, _Float16* __restrict__ Kh,
    _Float16* __restrict__ Vh) {
  extern __shared__ _Float16 Xlds[];  // 64 x XSTR halves
  const int tid = threadIdx.x;
  const int wave = tid >> 5;
  const int lane = tid & 31;
  const int h = lane >> 4;
  const int c = lane & 15;
  const int row0 = blockIdx.x * 64;

  // Stage x tile (fp32 global, coalesced float4) -> f16 LDS.
  for (int i = tid; i < 64 * 64; i += 256) {
    const int row = i >> 6, c4 = i & 63;
    const float4 f = *(const float4*)(x + (size_t)(row0 + row) * DD + c4 * 4);
    v4h hv;
    hv[0] = (_Float16)f.x; hv[1] = (_Float16)f.y;
    hv[2] = (_Float16)f.z; hv[3] = (_Float16)f.w;
    *(v4h*)(Xlds + (size_t)row * XSTR + c4 * 4) = hv;
  }
  __syncthreads();

  const int n0 = wave * 32;
  const _Float16* Ws[3] = {Wqh, Wkh, Wvh};
  const float* bs[3] = {bq, bk, bv};
  _Float16* outs[3] = {Qh, Kh, Vh};
  const float scales[3] = {0.0625f, 1.0f, 1.0f};  // fold 1/sqrt(D) into Q

#pragma unroll
  for (int w3 = 0; w3 < 3; ++w3) {
    const _Float16* W = Ws[w3];
    v8f acc[4][2];
#pragma unroll
    for (int mt = 0; mt < 4; ++mt) { acc[mt][0] = vzero8(); acc[mt][1] = vzero8(); }

    for (int kc = 0; kc < 8; ++kc) {
      // B-frags straight from global f16 weights (32B contiguous per lane;
      // weights live in L2 with massive reuse across 256 blocks).
      v16h b0 = load_b_frag(W + (size_t)n0 * DD, DD, kc * 32);
      v16h b1 = load_b_frag(W + (size_t)(n0 + 16) * DD, DD, kc * 32);
#pragma unroll
      for (int mt = 0; mt < 4; ++mt) {
        v16h a = load_a_frag(Xlds + (size_t)mt * 16 * XSTR, XSTR, kc * 32);
        acc[mt][0] = wmma16(a, b0, acc[mt][0]);
        acc[mt][1] = wmma16(a, b1, acc[mt][1]);
      }
    }

    const float sc = scales[w3];
    const float bias0 = bs[w3][n0 + c];
    const float bias1 = bs[w3][n0 + 16 + c];
    _Float16* op = outs[w3];
#pragma unroll
    for (int mt = 0; mt < 4; ++mt) {
#pragma unroll
      for (int r = 0; r < 8; ++r) {
        const size_t row = (size_t)(row0 + mt * 16 + r + 8 * h);
        op[row * DD + n0 + c]      = (_Float16)((acc[mt][0][r] + bias0) * sc);
        op[row * DD + n0 + 16 + c] = (_Float16)((acc[mt][1][r] + bias1) * sc);
      }
    }
  }
}

// ---- Kernel 2: flash attention. Block = 64 q rows (4 waves x 16), stream
// 64-key K/V tiles through LDS; online softmax; O accumulated in f32. -------
__global__ void __launch_bounds__(128) flash_attn(
    const _Float16* __restrict__ Qh, const _Float16* __restrict__ Kh,
    const _Float16* __restrict__ Vh, float* __restrict__ out) {
  extern __shared__ _Float16 smem[];
  _Float16* Klds = smem;                  // 64  x KSTR : K row-major
  _Float16* Vt   = Klds + 64 * KSTR;      // 256 x VSTR : V transposed (d-major)
  _Float16* Pall = Vt + 256 * VSTR;       // 4 waves x 16 x PSTR : P scratch

  const int tid = threadIdx.x;
  const int wave = tid >> 5;
  const int lane = tid & 31;
  const int h = lane >> 4;   // C-layout: lane holds rows r+8h, col c
  const int c = lane & 15;
  _Float16* Pw = Pall + (size_t)wave * 16 * PSTR;

  const int batch = blockIdx.y;
  const int qblk = blockIdx.x;
  const size_t bbase = (size_t)batch * SS * DD;
  const _Float16* Qg = Qh + bbase + (size_t)(qblk * 64 + wave * 16) * DD;
  const _Float16* Kg = Kh + bbase;
  const _Float16* Vg = Vh + bbase;

  // Q fragments resident in registers for the whole kernel (8 x v16h).
  v16h qf[8];
#pragma unroll
  for (int kc = 0; kc < 8; ++kc) qf[kc] = load_a_frag(Qg, DD, kc * 32);

  v8f oacc[16];
#pragma unroll
  for (int nt = 0; nt < 16; ++nt) oacc[nt] = vzero8();
  float m[8], l[8];
#pragma unroll
  for (int r = 0; r < 8; ++r) { m[r] = -3.0e38f; l[r] = 0.f; }

  const int ntiles = SS / 64;
  for (int kt = 0; kt < ntiles; ++kt) {
    const _Float16* Ksrc = Kg + (size_t)kt * 64 * DD;
    const _Float16* Vsrc = Vg + (size_t)kt * 64 * DD;
    if (kt + 1 < ntiles) {  // global_prefetch_b8 of next tile
      __builtin_prefetch(Kg + (size_t)(kt + 1) * 64 * DD + (size_t)tid * 128, 0, 1);
      __builtin_prefetch(Vg + (size_t)(kt + 1) * 64 * DD + (size_t)tid * 128, 0, 1);
    }

    // K tile: async DMA global->LDS (no VGPR round-trip, ASYNCcnt-tracked)
    // when available; otherwise staged through registers.
#ifdef HAVE_ASYNC_LDS
    for (int i = tid; i < 64 * 32; i += 128) {
      const int row = i >> 5, c8 = i & 31;
      async_ld_b128(Ksrc + (size_t)row * DD + c8 * 8,
                    Klds + (size_t)row * KSTR + c8 * 8);
    }
#else
    for (int i = tid; i < 64 * 32; i += 128) {
      const int row = i >> 5, c8 = i & 31;
      v8h kv = *(const v8h*)(Ksrc + (size_t)row * DD + c8 * 8);
      *(v8h*)(Klds + (size_t)row * KSTR + c8 * 8) = kv;
    }
#endif
    // V tile: transpose to d-major on the way into LDS (must pass through
    // VGPRs; makes P*V B-frags contiguous 32B reads).
    for (int i = tid; i < 64 * 32; i += 128) {
      const int row = i >> 5, c8 = i & 31;
      v8h vv = *(const v8h*)(Vsrc + (size_t)row * DD + c8 * 8);
#pragma unroll
      for (int e = 0; e < 8; ++e) Vt[(size_t)(c8 * 8 + e) * VSTR + row] = vv[e];
    }
#ifdef HAVE_ASYNC_LDS
    wait_async0();
#endif
    __syncthreads();

    // S = Q * K^T  (16 q-rows x 64 keys), 1/sqrt(D) already folded into Q.
    v8f s[4];
#pragma unroll
    for (int nt = 0; nt < 4; ++nt) s[nt] = vzero8();
    for (int kc = 0; kc < 8; ++kc) {
#pragma unroll
      for (int nt = 0; nt < 4; ++nt) {
        v16h b = load_b_frag(Klds + (size_t)nt * 16 * KSTR, KSTR, kc * 32);
        s[nt] = wmma16(qf[kc], b, s[nt]);
      }
    }

    // Online softmax. Each lane holds 8 rows x 4 cols; row stats reduced
    // across the 16-lane column group with shfl_xor (wave32).
    float rmax[8];
#pragma unroll
    for (int r = 0; r < 8; ++r)
      rmax[r] = fmaxf(fmaxf(s[0][r], s[1][r]), fmaxf(s[2][r], s[3][r]));
#pragma unroll
    for (int off = 1; off <= 8; off <<= 1) {
#pragma unroll
      for (int r = 0; r < 8; ++r)
        rmax[r] = fmaxf(rmax[r], __shfl_xor(rmax[r], off, 32));
    }

    float alpha[8], rsum[8];
#pragma unroll
    for (int r = 0; r < 8; ++r) {
      const float mn = fmaxf(m[r], rmax[r]);
      alpha[r] = __expf(m[r] - mn);
      m[r] = mn;
      rsum[r] = 0.f;
    }
#pragma unroll
    for (int nt = 0; nt < 4; ++nt) {
#pragma unroll
      for (int r = 0; r < 8; ++r) {
        const float p = __expf(s[nt][r] - m[r]);
        rsum[r] += p;
        Pw[(size_t)(r + 8 * h) * PSTR + nt * 16 + c] = (_Float16)p;
      }
    }
#pragma unroll
    for (int off = 1; off <= 8; off <<= 1) {
#pragma unroll
      for (int r = 0; r < 8; ++r) rsum[r] += __shfl_xor(rsum[r], off, 32);
    }
#pragma unroll
    for (int r = 0; r < 8; ++r) l[r] = l[r] * alpha[r] + rsum[r];
#pragma unroll
    for (int nt = 0; nt < 16; ++nt) {
#pragma unroll
      for (int r = 0; r < 8; ++r) oacc[nt][r] *= alpha[r];
    }

    // O += P * V  (DS ops from the same wave are in-order, so Pw write->read
    // within the wave needs no barrier.)
#pragma unroll
    for (int kk = 0; kk < 2; ++kk) {
      v16h pa = load_a_frag(Pw, PSTR, kk * 32);
#pragma unroll
      for (int nt = 0; nt < 16; ++nt) {
        v16h b = load_b_frag(Vt + (size_t)nt * 16 * VSTR, VSTR, kk * 32);
        oacc[nt] = wmma16(pa, b, oacc[nt]);
      }
    }
    __syncthreads();
  }

  // Normalize and write fp32 output.
  float* Og = out + bbase + (size_t)(qblk * 64 + wave * 16) * DD;
#pragma unroll
  for (int r = 0; r < 8; ++r) {
    const float inv = 1.f / l[r];
#pragma unroll
    for (int nt = 0; nt < 16; ++nt)
      Og[(size_t)(r + 8 * h) * DD + nt * 16 + c] = oacc[nt][r] * inv;
  }
}

// ---------------------------------------------------------------------------
extern "C" void kernel_launch(void* const* d_in, const int* in_sizes, int n_in,
                              void* d_out, int out_size, void* d_ws, size_t ws_size,
                              hipStream_t stream) {
  const float* x  = (const float*)d_in[0];
  const float* Wq = (const float*)d_in[1];
  const float* bq = (const float*)d_in[2];
  const float* Wk = (const float*)d_in[3];
  const float* bk = (const float*)d_in[4];
  const float* Wv = (const float*)d_in[5];
  const float* bv = (const float*)d_in[6];
  float* out = (float*)d_out;

  // Workspace layout (halves): 3 f16 weight copies + f16 Q,K,V (~24.4 MB).
  _Float16* wsh = (_Float16*)d_ws;
  _Float16* Wqh = wsh;
  _Float16* Wkh = Wqh + DD * DD;
  _Float16* Wvh = Wkh + DD * DD;
  _Float16* Qh  = Wvh + DD * DD;
  _Float16* Kh  = Qh + (size_t)BB * SS * DD;
  _Float16* Vh  = Kh + (size_t)BB * SS * DD;

  hipLaunchKernelGGL(cvt_weights, dim3(DD * DD / 256), dim3(256), 0, stream,
                     Wq, Wk, Wv, Wqh, Wkh, Wvh);

  const size_t xlds = (size_t)64 * XSTR * sizeof(_Float16);  // ~33.8 KB
  hipLaunchKernelGGL(qkv_proj, dim3(BB * SS / 64), dim3(256), xlds, stream,
                     x, Wqh, Wkh, Wvh, bq, bk, bv, Qh, Kh, Vh);

  const size_t flds =
      (size_t)(64 * KSTR + 256 * VSTR + 4 * 16 * PSTR) * sizeof(_Float16);  // ~78 KB
  hipLaunchKernelGGL(flash_attn, dim3(SS / 64, BB), dim3(128), flds, stream,
                     Qh, Kh, Vh, out);
}
